// LGSHead_274877907017
// MI455X (gfx1250) — matrix operand order repaired and checked by reading, not compiled, not run on hardware
//
#include <hip/hip_runtime.h>
#include <hip/hip_bf16.h>

typedef __attribute__((ext_vector_type(16))) _Float16 v16h;
typedef __attribute__((ext_vector_type(8)))  _Float16 v8h;
typedef __attribute__((ext_vector_type(8)))  float    v8f;

__device__ __forceinline__ float preluf(float x, float a) { return x >= 0.0f ? x : a * x; }

// A-fragment (16x32 f16, M x K) per CDNA5 layout:
// lane L -> row M (caller picks via rowp), halves 0..7 = K base..base+7,
// halves 8..15 = K base+16..base+23, base = 8*(lane>=16).
__device__ __forceinline__ v16h load_a_frag(const _Float16* __restrict__ rowp, int lane) {
  const int base = (lane >> 4) << 3;
  v8h lo = *(const v8h*)(rowp + base);
  v8h hi = *(const v8h*)(rowp + base + 16);
  v16h r;
#pragma unroll
  for (int i = 0; i < 8; ++i) { r[i] = lo[i]; r[i + 8] = hi[i]; }
  return r;
}

#define WMMA_F16(a, b, c) \
  __builtin_amdgcn_wmma_f32_16x16x32_f16(false, (a), false, (b), (short)0, (c), false, false)

// ---------------------------------------------------------------------------
// GroupNorm(16 groups over 768 ch, 256 patches) + affine + PReLU -> f16 [3072][256]
// mins layout: [B=4][p=256][c=768]
// ---------------------------------------------------------------------------
__global__ __launch_bounds__(256) void gn_prelu_kernel(
    const float* __restrict__ mins, const float* __restrict__ gamma,
    const float* __restrict__ beta, const float* __restrict__ gap_p,
    _Float16* __restrict__ xfc) {
  __shared__ float r1[256], r2[256];
  const int bg = blockIdx.x;              // 64 blocks = B*16 groups
  const int b = bg >> 4, g = bg & 15;
  const int c0 = g * 48;
  const int tid = threadIdx.x;
  float s = 0.0f, ss = 0.0f;
  for (int e = tid; e < 48 * 256; e += 256) {
    int c = c0 + (e >> 8), p = e & 255;
    float v = mins[((size_t)b * 256 + p) * 768 + c];
    s += v; ss += v * v;
  }
  r1[tid] = s; r2[tid] = ss; __syncthreads();
  for (int st = 128; st > 0; st >>= 1) {
    if (tid < st) { r1[tid] += r1[tid + st]; r2[tid] += r2[tid + st]; }
    __syncthreads();
  }
  const float invN = 1.0f / (48.0f * 256.0f);
  const float mean = r1[0] * invN;
  const float var  = r2[0] * invN - mean * mean;
  const float rstd = rsqrtf(var + 1e-5f);
  const float a = *gap_p;
  for (int e = tid; e < 48 * 256; e += 256) {
    int c = c0 + (e >> 8), p = e & 255;
    float v = mins[((size_t)b * 256 + p) * 768 + c];
    float y = (v - mean) * rstd * gamma[c] + beta[c];
    xfc[((size_t)b * 768 + c) * 256 + p] = (_Float16)preluf(y, a);
  }
}

// ---------------------------------------------------------------------------
// Transpose fp32 [N][K] weight -> f16 [K][N]
// ---------------------------------------------------------------------------
__global__ void transpose_h_kernel(const float* __restrict__ W, _Float16* __restrict__ out,
                                   int Nrows, int Kcols) {
  int t = blockIdx.x * blockDim.x + threadIdx.x;
  if (t >= Nrows * Kcols) return;
  int n = t / Kcols, k = t % Kcols;
  out[(size_t)k * Nrows + n] = (_Float16)W[t];
}

// Plain fp32 -> f16 convert (for the conv1 input image)
__global__ void f32_to_f16_kernel(const float* __restrict__ in, _Float16* __restrict__ out,
                                  int n) {
  int t = blockIdx.x * blockDim.x + threadIdx.x;
  if (t < n) out[t] = (_Float16)in[t];
}

// ---------------------------------------------------------------------------
// tf = concat([text @ bg_W^T + bg_b, text], axis=1) -> f32 [4][8][512]
// ---------------------------------------------------------------------------
__global__ void tf_kernel(const float* __restrict__ text, const float* __restrict__ bgW,
                          const float* __restrict__ bgb, float* __restrict__ tf) {
  int t = blockIdx.x * blockDim.x + threadIdx.x;     // 16384
  if (t >= 4 * 8 * 512) return;
  int d = t & 511, i = (t >> 9) & 7, b = t >> 12;
  float v;
  if (i >= 4) {
    v = text[((size_t)b * 4 + (i - 4)) * 512 + d];
  } else {
    const float* tr = text + ((size_t)b * 4 + i) * 512;
    const float* wr = bgW + (size_t)d * 512;
    float acc = bgb[d];
    for (int k = 0; k < 512; ++k) acc += tr[k] * wr[k];
    v = acc;
  }
  tf[t] = v;
}

// ---------------------------------------------------------------------------
// WMMA GEMM: C[M][N] (f16) = optional_prelu(A[M][K] f16 @ B[K][N] f16 + bias[N])
// Block: 256 thr = 8 waves -> 128 x 64 output tile (wave = 16 rows, 4 N-subtiles)
// ---------------------------------------------------------------------------
__global__ __launch_bounds__(256) void gemm_wmma_kernel(
    const _Float16* __restrict__ A, const _Float16* __restrict__ Bm,
    const float* __restrict__ bias, _Float16* __restrict__ C,
    int M, int N, int K, const float* __restrict__ slope_p, int do_prelu) {
  const int tid = threadIdx.x, lane = tid & 31, wave = tid >> 5;
  const int m0 = blockIdx.y * 128 + wave * 16;
  const int n0 = blockIdx.x * 64;
  v8f acc0 = {}, acc1 = {}, acc2 = {}, acc3 = {};
  const _Float16* arow = A + (size_t)(m0 + (lane & 15)) * K;
  for (int k0 = 0; k0 < K; k0 += 32) {
    v16h a = load_a_frag(arow + k0, lane);
    const _Float16* brow = Bm + (size_t)k0 * N + n0 + (lane & 15);   // lane -> N column
    __builtin_prefetch(brow + (size_t)32 * N, 0, 1);            // global_prefetch_b8
    v16h b0, b1, b2, b3;
#pragma unroll
    for (int i = 0; i < 16; ++i) {
      int kr = ((lane >> 4) << 3) + (i & 7) + ((i >> 3) << 4);
      const _Float16* bq = brow + (size_t)kr * N;
      b0[i] = bq[0]; b1[i] = bq[16]; b2[i] = bq[32]; b3[i] = bq[48];
    }
    acc0 = WMMA_F16(a, b0, acc0);
    acc1 = WMMA_F16(a, b1, acc1);
    acc2 = WMMA_F16(a, b2, acc2);
    acc3 = WMMA_F16(a, b3, acc3);
  }
  const float slope = do_prelu ? *slope_p : 0.0f;
  const int nc = lane & 15, mh = (lane >> 4) << 3;
  v8f accs[4] = {acc0, acc1, acc2, acc3};
#pragma unroll
  for (int nt = 0; nt < 4; ++nt) {
    int n = n0 + nt * 16 + nc;
    float bv = bias[n];
#pragma unroll
    for (int r = 0; r < 8; ++r) {
      float v = accs[nt][r] + bv;
      if (do_prelu) v = preluf(v, slope);
      C[(size_t)(m0 + r + mh) * N + n] = (_Float16)v;
    }
  }
}

// ---------------------------------------------------------------------------
// q projection (tiny, M=32): f32 VALU linear
// ---------------------------------------------------------------------------
__global__ void linear_f32_kernel(const float* __restrict__ X, const float* __restrict__ W,
                                  const float* __restrict__ bvec, float* __restrict__ Y,
                                  int M, int N, int K) {
  int t = blockIdx.x * blockDim.x + threadIdx.x;
  if (t >= M * N) return;
  int n = t % N, m = t / N;
  const float* xr = X + (size_t)m * K;
  const float* wr = W + (size_t)n * K;
  float acc = bvec[n];
  for (int k = 0; k < K; ++k) acc += xr[k] * wr[k];
  Y[t] = acc;
}

// ---------------------------------------------------------------------------
// Attention for one (b, head): scores [8][768], softmax, weighted sum of V.
// ---------------------------------------------------------------------------
__global__ __launch_bounds__(256) void attn_kernel(
    const float* __restrict__ qh, const _Float16* __restrict__ kh,
    const _Float16* __restrict__ vh, float* __restrict__ oin) {
  __shared__ float sbuf[8 * 768];
  __shared__ float red[256];
  __shared__ float qs[8 * 64];
  const int bh = blockIdx.x;      // 32 = B * nh
  const int b = bh >> 3, h = bh & 7;
  const int tid = threadIdx.x;
  for (int e = tid; e < 512; e += 256) {
    int i = e >> 6, dd = e & 63;
    qs[e] = qh[((size_t)b * 8 + i) * 512 + h * 64 + dd] * 0.125f;  // 1/sqrt(64)
  }
  __syncthreads();
  for (int j = tid; j < 768; j += 256) {
    const _Float16* kr = kh + ((size_t)b * 768 + j) * 512 + h * 64;
    for (int i = 0; i < 8; ++i) {
      float acc = 0.0f;
      for (int dd = 0; dd < 64; ++dd) acc += qs[i * 64 + dd] * (float)kr[dd];
      sbuf[i * 768 + j] = acc;
    }
  }
  __syncthreads();
  for (int i = 0; i < 8; ++i) {
    float lm = -1e30f;
    for (int j = tid; j < 768; j += 256) lm = fmaxf(lm, sbuf[i * 768 + j]);
    red[tid] = lm; __syncthreads();
    for (int st = 128; st > 0; st >>= 1) {
      if (tid < st) red[tid] = fmaxf(red[tid], red[tid + st]);
      __syncthreads();
    }
    float mx = red[0]; __syncthreads();
    float ls = 0.0f;
    for (int j = tid; j < 768; j += 256) {
      float e = __expf(sbuf[i * 768 + j] - mx);
      sbuf[i * 768 + j] = e; ls += e;
    }
    red[tid] = ls; __syncthreads();
    for (int st = 128; st > 0; st >>= 1) {
      if (tid < st) red[tid] += red[tid + st];
      __syncthreads();
    }
    float inv = 1.0f / red[0]; __syncthreads();
    for (int j = tid; j < 768; j += 256) sbuf[i * 768 + j] *= inv;
    __syncthreads();
  }
  for (int e = tid; e < 512; e += 256) {
    int i = e >> 6, dd = e & 63;
    const _Float16* vcol = vh + (size_t)b * 768 * 512 + h * 64 + dd;
    float acc = 0.0f;
    for (int j = 0; j < 768; ++j) acc += sbuf[i * 768 + j] * (float)vcol[(size_t)j * 512];
    oin[((size_t)b * 8 + i) * 512 + h * 64 + dd] = acc;
  }
}

// ---------------------------------------------------------------------------
// cross = prelu(prelu(oin @ o_W^T + o_b, mha) + tf, ca)   -> f32 [4][8][512]
// ---------------------------------------------------------------------------
__global__ void cross_kernel(const float* __restrict__ oin, const float* __restrict__ oW,
                             const float* __restrict__ ob, const float* __restrict__ tf,
                             const float* __restrict__ mha_p, const float* __restrict__ ca_p,
                             float* __restrict__ cross) {
  int t = blockIdx.x * blockDim.x + threadIdx.x;
  if (t >= 16384) return;
  int d = t & 511, m = t >> 9;
  const float* xr = oin + (size_t)m * 512;
  const float* wr = oW + (size_t)d * 512;
  float acc = ob[d];
  for (int k = 0; k < 512; ++k) acc += xr[k] * wr[k];
  acc = preluf(acc, *mha_p);
  acc = preluf(acc + tf[t], *ca_p);
  cross[t] = acc;
}

// rw_flat[s = q*B + b][e] = sigmoid(cross[b][q] . rt_W[e] + rt_b[e])
__global__ void rw_kernel(const float* __restrict__ cross, const float* __restrict__ rtW,
                          const float* __restrict__ rtb, float* __restrict__ rw_flat) {
  int t = threadIdx.x;                 // exactly 256 = 4*8*8
  int e = t & 7, q = (t >> 3) & 7, b = t >> 6;
  const float* xr = cross + ((size_t)b * 8 + q) * 512;
  const float* wr = rtW + (size_t)e * 512;
  float acc = rtb[e];
  for (int k = 0; k < 512; ++k) acc += xr[k] * wr[k];
  rw_flat[((size_t)q * 4 + b) * 8 + e] = 1.0f / (1.0f + __expf(-acc));
}

// Mix 8 expert kernels per sample: out[s][j] = sum_e coef[s][e] * Wexp[e][j]  (f16 out)
__global__ void mix_kernel(const float* __restrict__ coef, const float* __restrict__ Wexp,
                           _Float16* __restrict__ out, int J) {
  int t = blockIdx.x * blockDim.x + threadIdx.x;
  if (t >= 32 * J) return;
  int s = t / J, j = t % J;
  const float* c = coef + (size_t)s * 8;
  float acc = 0.0f;
#pragma unroll
  for (int e = 0; e < 8; ++e) acc += c[e] * Wexp[(size_t)e * J + j];
  out[t] = (_Float16)acc;
}

// ---------------------------------------------------------------------------
// CondConv1 (implicit GEMM, WMMA): per sample s, out[128][128][128] =
//   prelu(K1[s] (128x576) @ im2col(input_h[b]) ) -> xc1 f16 [32][128][16384]
// Block: 8 waves, one row y, 64-px strip; halo [64][3][66] f16 in LDS.
// ---------------------------------------------------------------------------
__global__ __launch_bounds__(256) void conv1_wmma(
    const _Float16* __restrict__ x, const _Float16* __restrict__ K1,
    _Float16* __restrict__ xc1, const float* __restrict__ slope_p) {
  __shared__ _Float16 hal[64 * 3 * 66];
  const int s = blockIdx.z, b = s & 3;
  const int y = blockIdx.y, x0 = blockIdx.x * 64;
  const int tid = threadIdx.x, lane = tid & 31, wave = tid >> 5;
  for (int e = tid; e < 64 * 198; e += 256) {
    int ic = e / 198, r = e % 198, dy = r / 66, xx = r % 66;
    int yy = y + dy - 1, gx = x0 + xx - 1;
    _Float16 v = (_Float16)0.0f;
    if (yy >= 0 && yy < 128 && gx >= 0 && gx < 128)
      v = x[(((size_t)b * 64 + ic) * 128 + yy) * 128 + gx];
    hal[e] = v;
  }
  __syncthreads();
  const int m0 = wave * 16;
  const _Float16* arow = K1 + (size_t)s * 73728 + (size_t)(m0 + (lane & 15)) * 576;
  v8f acc0 = {}, acc1 = {}, acc2 = {}, acc3 = {};
  for (int k0 = 0; k0 < 576; k0 += 32) {
    v16h a = load_a_frag(arow + k0, lane);
    v16h b0, b1, b2, b3;
#pragma unroll
    for (int i = 0; i < 16; ++i) {
      int kk = k0 + ((lane >> 4) << 3) + (i & 7) + ((i >> 3) << 4);
      int ic = kk / 9, off = kk % 9, dy = off / 3, dx = off % 3;
      const _Float16* bp = &hal[(ic * 3 + dy) * 66 + dx + (lane & 15)];
      b0[i] = bp[0]; b1[i] = bp[16]; b2[i] = bp[32]; b3[i] = bp[48];
    }
    acc0 = WMMA_F16(a, b0, acc0);
    acc1 = WMMA_F16(a, b1, acc1);
    acc2 = WMMA_F16(a, b2, acc2);
    acc3 = WMMA_F16(a, b3, acc3);
  }
  const float slope = *slope_p;
  const int nc = lane & 15, mh = (lane >> 4) << 3;
  v8f accs[4] = {acc0, acc1, acc2, acc3};
#pragma unroll
  for (int nt = 0; nt < 4; ++nt) {
#pragma unroll
    for (int r = 0; r < 8; ++r) {
      int m = m0 + r + mh;
      xc1[((size_t)s * 128 + m) * 16384 + y * 128 + x0 + nt * 16 + nc] =
          (_Float16)preluf(accs[nt][r], slope);
    }
  }
}

// ---------------------------------------------------------------------------
// pooled[s][oc] = mean over 16384 px of xc1[s][oc].
// Streams the 32KB channel plane into LDS via GLOBAL_LOAD_ASYNC_TO_LDS_B128
// (ASYNCcnt path, no VGPR round-trip), then reduces out of LDS.
// ---------------------------------------------------------------------------
__global__ __launch_bounds__(256) void pool_kernel(const _Float16* __restrict__ xc1,
                                                   float* __restrict__ pooled) {
  __shared__ _Float16 buf[16384];
  __shared__ float red[256];
  const int blk = blockIdx.x;          // 4096 = 32*128
  const _Float16* base = xc1 + (size_t)blk * 16384;
  const int tid = threadIdx.x;
  // Per-lane LDS byte offset (low 32 bits of the generic pointer are the LDS addr)
  unsigned lds0 = (unsigned)(uintptr_t)(&buf[0]) + (unsigned)tid * 16u;
#pragma unroll
  for (int it = 0; it < 8; ++it) {
    const _Float16* g = base + it * 2048 + tid * 8;   // 16B per lane, 16B-aligned
    unsigned l = lds0 + (unsigned)it * 4096u;
    asm volatile("global_load_async_to_lds_b128 %0, %1, off"
                 :: "v"(l), "v"(g) : "memory");
  }
  asm volatile("s_wait_asynccnt 0x0" ::: "memory");
  __syncthreads();
  float s = 0.0f;
  for (int e = tid; e < 16384; e += 256) s += (float)buf[e];
  red[tid] = s; __syncthreads();
  for (int st = 128; st > 0; st >>= 1) {
    if (tid < st) red[tid] += red[tid + st];
    __syncthreads();
  }
  if (tid == 0) pooled[blk] = red[0] * (1.0f / 16384.0f);
}

// sr[s][e] = pooled[s] . sr_W[e] + sr_b[e]   (no sigmoid)
__global__ void sr_kernel(const float* __restrict__ pooled, const float* __restrict__ srW,
                          const float* __restrict__ srb, float* __restrict__ sr) {
  int t = threadIdx.x;                 // 256 = 32*8
  int e = t & 7, s = t >> 3;
  const float* pr = pooled + (size_t)s * 128;
  const float* wr = srW + (size_t)e * 128;
  float acc = srb[e];
  for (int k = 0; k < 128; ++k) acc += pr[k] * wr[k];
  sr[t] = acc;
}

// ---------------------------------------------------------------------------
// CondConv2 (WMMA) fused with prelu + co 1x1 projection + channel reduce:
//   logits[s][p] = sum_oc co_W[oc]*prelu(K2[s] @ im2col(xc1[s]))[oc][p] + co_b
// ---------------------------------------------------------------------------
__global__ __launch_bounds__(256) void conv2_wmma(
    const _Float16* __restrict__ xc1, const _Float16* __restrict__ K2,
    const float* __restrict__ coW, const float* __restrict__ coB,
    float* __restrict__ logits, const float* __restrict__ slope_p) {
  __shared__ _Float16 hal[128 * 3 * 66];
  __shared__ float red[4][64];
  const int s = blockIdx.z;
  const int y = blockIdx.y, x0 = blockIdx.x * 64;
  const int tid = threadIdx.x, lane = tid & 31, wave = tid >> 5;
  for (int e = tid; e < 128 * 198; e += 256) {
    int ic = e / 198, r = e % 198, dy = r / 66, xx = r % 66;
    int yy = y + dy - 1, gx = x0 + xx - 1;
    _Float16 v = (_Float16)0.0f;
    if (yy >= 0 && yy < 128 && gx >= 0 && gx < 128)
      v = xc1[((size_t)s * 128 + ic) * 16384 + yy * 128 + gx];
    hal[e] = v;
  }
  __syncthreads();
  const int mt = wave >> 1, ntb = (wave & 1) * 2;
  const int m0 = mt * 16;
  const _Float16* arow = K2 + (size_t)s * 73728 + (size_t)(m0 + (lane & 15)) * 1152;
  v8f acc0 = {}, acc1 = {};
  for (int k0 = 0; k0 < 1152; k0 += 32) {
    v16h a = load_a_frag(arow + k0, lane);
    v16h b0, b1;
#pragma unroll
    for (int i = 0; i < 16; ++i) {
      int kk = k0 + ((lane >> 4) << 3) + (i & 7) + ((i >> 3) << 4);
      int ic = kk / 9, off = kk % 9, dy = off / 3, dx = off % 3;
      const _Float16* bp = &hal[(ic * 3 + dy) * 66 + dx + ntb * 16 + (lane & 15)];
      b0[i] = bp[0]; b1[i] = bp[16];
    }
    acc0 = WMMA_F16(a, b0, acc0);
    acc1 = WMMA_F16(a, b1, acc1);
  }
  const float slope = *slope_p;
  const int nc = lane & 15, mh = (lane >> 4) << 3;
  float ps0 = 0.0f, ps1 = 0.0f;
#pragma unroll
  for (int r = 0; r < 8; ++r) {
    float cw = coW[m0 + r + mh];
    ps0 += preluf(acc0[r], slope) * cw;
    ps1 += preluf(acc1[r], slope) * cw;
  }
  ps0 += __shfl_xor(ps0, 16, 32);     // combine m-halves (lanes L and L^16 share n)
  ps1 += __shfl_xor(ps1, 16, 32);
  if (lane < 16) {
    red[mt][ntb * 16 + nc] = ps0;
    red[mt][(ntb + 1) * 16 + nc] = ps1;
  }
  __syncthreads();
  if (tid < 64) {
    float v = red[0][tid] + red[1][tid] + red[2][tid] + red[3][tid] + coB[0];
    logits[(size_t)s * 16384 + y * 128 + x0 + tid] = v;
  }
}

// softmax over the 8 queries; out[b][q][y][x] = softmax_q(logits[q*4+b][p])
__global__ void softmax_out_kernel(const float* __restrict__ logits, float* __restrict__ out) {
  int t = blockIdx.x * blockDim.x + threadIdx.x;     // 65536
  if (t >= 65536) return;
  int p = t & 16383, b = t >> 14;
  float v[8]; float mx = -1e30f;
#pragma unroll
  for (int q = 0; q < 8; ++q) {
    v[q] = logits[((size_t)q * 4 + b) * 16384 + p];
    mx = fmaxf(mx, v[q]);
  }
  float sum = 0.0f;
#pragma unroll
  for (int q = 0; q < 8; ++q) { v[q] = __expf(v[q] - mx); sum += v[q]; }
  float inv = 1.0f / sum;
#pragma unroll
  for (int q = 0; q < 8; ++q) out[((size_t)b * 8 + q) * 16384 + p] = v[q] * inv;
}

__global__ void copy_tf_kernel(const float* __restrict__ tf, float* __restrict__ out) {
  int t = blockIdx.x * blockDim.x + threadIdx.x;
  if (t < 16384) out[524288 + t] = tf[t];
}

// ---------------------------------------------------------------------------
extern "C" void kernel_launch(void* const* d_in, const int* in_sizes, int n_in,
                              void* d_out, int out_size, void* d_ws, size_t ws_size,
                              hipStream_t stream) {
  (void)in_sizes; (void)n_in; (void)out_size; (void)ws_size;
  const float* mins  = (const float*)d_in[0];
  const float* input = (const float*)d_in[1];
  const float* text  = (const float*)d_in[2];
  const float* gamma = (const float*)d_in[3];
  const float* beta  = (const float*)d_in[4];
  const float* gap   = (const float*)d_in[5];
  const float* bgW   = (const float*)d_in[6];
  const float* bgb   = (const float*)d_in[7];
  const float* lbW   = (const float*)d_in[8];
  const float* lbb   = (const float*)d_in[9];
  const float* ca    = (const float*)d_in[10];
  const float* qW    = (const float*)d_in[11];
  const float* qb    = (const float*)d_in[12];
  const float* kW    = (const float*)d_in[13];
  const float* kb    = (const float*)d_in[14];
  const float* vW    = (const float*)d_in[15];
  const float* vb    = (const float*)d_in[16];
  const float* oW    = (const float*)d_in[17];
  const float* ob    = (const float*)d_in[18];
  const float* mha_s = (const float*)d_in[19];
  const float* rtW   = (const float*)d_in[20];
  const float* rtb   = (const float*)d_in[21];
  const float* cc1   = (const float*)d_in[22];
  const float* cc2   = (const float*)d_in[23];
  const float* srW   = (const float*)d_in[24];
  const float* srb   = (const float*)d_in[25];
  const float* coW   = (const float*)d_in[26];
  const float* cob   = (const float*)d_in[27];
  const float* dc    = (const float*)d_in[28];
  float* out = (float*)d_out;

  char* w = (char*)d_ws;
  size_t off = 0;
  auto alloc = [&](size_t bytes) -> void* {
    void* p = w + off;
    off = (off + bytes + 255) & ~(size_t)255;
    return p;
  };
  _Float16* xfc    = (_Float16*)alloc((size_t)3072 * 256 * 2);
  _Float16* lbT    = (_Float16*)alloc((size_t)256 * 512 * 2);
  _Float16* kT     = (_Float16*)alloc((size_t)512 * 512 * 2);
  _Float16* vT     = (_Float16*)alloc((size_t)512 * 512 * 2);
  _Float16* key_h  = (_Float16*)alloc((size_t)3072 * 512 * 2);
  _Float16* kh_h   = (_Float16*)alloc((size_t)3072 * 512 * 2);
  _Float16* vh_h   = (_Float16*)alloc((size_t)3072 * 512 * 2);
  float*    tf     = (float*)alloc((size_t)16384 * 4);
  float*    qh     = (float*)alloc((size_t)16384 * 4);
  float*    oin    = (float*)alloc((size_t)16384 * 4);
  float*    cross  = (float*)alloc((size_t)16384 * 4);
  float*    rwf    = (float*)alloc((size_t)256 * 4);
  _Float16* inp_h  = (_Float16*)alloc((size_t)4 * 64 * 128 * 128 * 2);
  _Float16* K1     = (_Float16*)alloc((size_t)32 * 73728 * 2);
  _Float16* xc1    = (_Float16*)alloc((size_t)32 * 128 * 16384 * 2);
  float*    pooled = (float*)alloc((size_t)4096 * 4);
  float*    sr     = (float*)alloc((size_t)256 * 4);
  _Float16* K2     = (_Float16*)alloc((size_t)32 * 73728 * 2);
  float*    logits = (float*)alloc((size_t)32 * 16384 * 4);

  // Stage 1: GroupNorm + PReLU -> x_feat_cat f16
  gn_prelu_kernel<<<64, 256, 0, stream>>>(mins, gamma, beta, gap, xfc);
  // Weight transposes (f32 [N][K] -> f16 [K][N]) for WMMA B-operands
  transpose_h_kernel<<<(512 * 256 + 255) / 256, 256, 0, stream>>>(lbW, lbT, 512, 256);
  transpose_h_kernel<<<(512 * 512 + 255) / 256, 256, 0, stream>>>(kW, kT, 512, 512);
  transpose_h_kernel<<<(512 * 512 + 255) / 256, 256, 0, stream>>>(vW, vT, 512, 512);
  // f16 copy of the conv input image (halves conv1 halo traffic)
  f32_to_f16_kernel<<<(4 * 64 * 128 * 128) / 256, 256, 0, stream>>>(input, inp_h,
                                                                    4 * 64 * 128 * 128);
  // tf = [bg ; text]
  tf_kernel<<<64, 256, 0, stream>>>(text, bgW, bgb, tf);
  // key_f = prelu(xfc @ lb_W^T + lb_b)  (WMMA, M=3072 K=256 N=512)
  gemm_wmma_kernel<<<dim3(8, 24), 256, 0, stream>>>(xfc, lbT, lbb, key_h, 3072, 512, 256, ca, 1);
  // k/v projections (WMMA, M=3072 K=512 N=512)
  gemm_wmma_kernel<<<dim3(8, 24), 256, 0, stream>>>(key_h, kT, kb, kh_h, 3072, 512, 512, ca, 0);
  gemm_wmma_kernel<<<dim3(8, 24), 256, 0, stream>>>(key_h, vT, vb, vh_h, 3072, 512, 512, ca, 0);
  // q projection (tiny) + attention + output proj/residual
  linear_f32_kernel<<<64, 256, 0, stream>>>(tf, qW, qb, qh, 32, 512, 512);
  attn_kernel<<<32, 256, 0, stream>>>(qh, kh_h, vh_h, oin);
  cross_kernel<<<64, 256, 0, stream>>>(oin, oW, ob, tf, mha_s, ca, cross);
  // routing weights + expert-mixed conv1 kernels
  rw_kernel<<<1, 256, 0, stream>>>(cross, rtW, rtb, rwf);
  mix_kernel<<<(32 * 73728) / 256, 256, 0, stream>>>(rwf, cc1, K1, 73728);
  // CondConv1 (WMMA implicit GEMM) -> xc1
  conv1_wmma<<<dim3(2, 128, 32), 256, 0, stream>>>(inp_h, K1, xc1, dc);
  // global average pool (async-to-LDS), self-routing, conv2 kernels
  pool_kernel<<<4096, 256, 0, stream>>>(xc1, pooled);
  sr_kernel<<<1, 256, 0, stream>>>(pooled, srW, srb, sr);
  mix_kernel<<<(32 * 73728) / 256, 256, 0, stream>>>(sr, cc2, K2, 73728);
  // CondConv2 (WMMA) fused with prelu + co projection + channel reduction
  conv2_wmma<<<dim3(2, 128, 32), 256, 0, stream>>>(xc1, K2, coW, cob, logits, dc);
  // softmax over queries + tf passthrough
  softmax_out_kernel<<<256, 256, 0, stream>>>(logits, out);
  copy_tf_kernel<<<64, 256, 0, stream>>>(tf, out);
}
